// NoisyLayer_35820027249485
// MI455X (gfx1250) — matrix operand order repaired and checked by reading, not compiled, not run on hardware
//
#include <hip/hip_runtime.h>

// NoisyNet dense layer on gfx1250 (MI455X):
//   out = x @ Wmu + eps_out * ((x*eps_in) @ Wsig + bias_sigma) + bias_mu
// Two fused f16 WMMA GEMMs with f32 accumulation, LDS-staged tiles.

typedef __attribute__((ext_vector_type(16))) _Float16 v16h;
typedef __attribute__((ext_vector_type(8)))  _Float16 h8;
typedef __attribute__((ext_vector_type(8)))  float    v8f;
typedef __attribute__((ext_vector_type(4)))  float    f32x4;

#define TILE_M 128
#define TILE_N 128
#define TILE_K 32
#define LDS_STRIDE (TILE_K + 8)   // 40 halves = 80B row stride; 16B aligned, conflict-breaking

union H16 { v16h v; h8 h[2]; };

__global__ __launch_bounds__(256)
void noisy_dense_wmma(const float* __restrict__ x,
                      const float* __restrict__ wmu,
                      const float* __restrict__ wsg,
                      const float* __restrict__ bmu,
                      const float* __restrict__ bsg,
                      const float* __restrict__ eps_in,
                      const float* __restrict__ eps_out,
                      float* __restrict__ out,
                      int Ddim, int Udim)
{
    __shared__ _Float16 sAx[TILE_M][LDS_STRIDE];   // f16(x) tile           [m][k]
    __shared__ _Float16 sAn[TILE_M][LDS_STRIDE];   // f16(x*eps_in) tile    [m][k]
    __shared__ _Float16 sBm[TILE_N][LDS_STRIDE];   // f16(Wmu) transposed   [n][k]
    __shared__ _Float16 sBs[TILE_N][LDS_STRIDE];   // f16(Wsig) transposed  [n][k]

    const int tid  = threadIdx.x;
    const int lane = tid & 31;
    const int wid  = tid >> 5;
    const int wm   = wid >> 1;        // 0..3 : 32-row slab of the block tile
    const int wn   = wid & 1;         // 0..1 : 64-col slab

    const int m0 = blockIdx.y * TILE_M;
    const int n0 = blockIdx.x * TILE_N;

    v8f accMu[2][4] = {};             // mean GEMM accumulators
    v8f accNz[2][4] = {};             // noise GEMM accumulators

    // WMMA fragment addressing (wave32 layouts per CDNA5 ISA 7.12.2)
    const int mlo  = lane & 15;       // row (A) / col (B,C,D) within 16
    const int ksel = lane >> 4;       // lane-group select
    const int khiA = ksel << 3;       // A frag: K base 0 or 8 (chunks at +0 and +16)
    const int kbB  = ksel << 3;       // B frag: K base 0 or 8 (chunks at +0 and +16)

    for (int k0 = 0; k0 < Ddim; k0 += TILE_K) {
        // ---- stage A tiles: convert f32 -> f16, build x and x*eps_in ----
        #pragma unroll
        for (int p = 0; p < 4; ++p) {
            int i  = tid + p * 256;           // 0..1023 float4 slots
            int r  = i >> 3;                  // 0..127 row
            int kc = (i & 7) << 2;            // 0..28 col base
            size_t g = (size_t)(m0 + r) * Ddim + k0 + kc;
            f32x4 xv = *(const f32x4*)(x + g);
            f32x4 ev = *(const f32x4*)(eps_in + g);
            #pragma unroll
            for (int j = 0; j < 4; ++j) {
                sAx[r][kc + j] = (_Float16)xv[j];
                sAn[r][kc + j] = (_Float16)(xv[j] * ev[j]);
            }
        }
        // ---- stage B tiles transposed: sB[n][k] ----
        #pragma unroll
        for (int p = 0; p < 4; ++p) {
            int i  = tid + p * 256;
            int kk = i >> 5;                  // 0..31 k row
            int nc = (i & 31) << 2;           // 0..124 n base
            size_t g = (size_t)(k0 + kk) * Udim + n0 + nc;
            f32x4 wmv = *(const f32x4*)(wmu + g);
            f32x4 wsv = *(const f32x4*)(wsg + g);
            #pragma unroll
            for (int j = 0; j < 4; ++j) {
                sBm[nc + j][kk] = (_Float16)wmv[j];
                sBs[nc + j][kk] = (_Float16)wsv[j];
            }
        }
        __syncthreads();

        // hint next K tiles toward the caches (global_prefetch_b8)
        if (k0 + TILE_K < Ddim) {
            int r  = tid >> 3;  int kc = (tid & 7) << 2;
            int kk = tid >> 5;  int nc = (tid & 31) << 2;
            __builtin_prefetch(x      + (size_t)(m0 + r) * Ddim + k0 + TILE_K + kc, 0, 1);
            __builtin_prefetch(eps_in + (size_t)(m0 + r) * Ddim + k0 + TILE_K + kc, 0, 1);
            __builtin_prefetch(wmu + (size_t)(k0 + TILE_K + kk) * Udim + n0 + nc, 0, 1);
            __builtin_prefetch(wsg + (size_t)(k0 + TILE_K + kk) * Udim + n0 + nc, 0, 1);
        }

        // ---- load A fragments (resident across the tn loop) ----
        H16 aX[2], aN[2];
        #pragma unroll
        for (int tm = 0; tm < 2; ++tm) {
            int row = wm * 32 + tm * 16 + mlo;
            aX[tm].h[0] = *(const h8*)&sAx[row][khiA];        // K = khiA..khiA+7
            aX[tm].h[1] = *(const h8*)&sAx[row][16 + khiA];   // K = 16+khiA..+7
            aN[tm].h[0] = *(const h8*)&sAn[row][khiA];
            aN[tm].h[1] = *(const h8*)&sAn[row][16 + khiA];
        }
        // ---- B fragments + WMMA ----
        #pragma unroll
        for (int tn = 0; tn < 4; ++tn) {
            int col = wn * 64 + tn * 16 + mlo;
            H16 bM, bS;
            bM.h[0] = *(const h8*)&sBm[col][kbB];             // K = kbB..kbB+7
            bM.h[1] = *(const h8*)&sBm[col][kbB + 16];        // K = 16+kbB..+7
            bS.h[0] = *(const h8*)&sBs[col][kbB];
            bS.h[1] = *(const h8*)&sBs[col][kbB + 16];
            #pragma unroll
            for (int tm = 0; tm < 2; ++tm) {
                accMu[tm][tn] = __builtin_amdgcn_wmma_f32_16x16x32_f16(
                    false, aX[tm].v, false, bM.v, (short)0, accMu[tm][tn], false, false);
                accNz[tm][tn] = __builtin_amdgcn_wmma_f32_16x16x32_f16(
                    false, aN[tm].v, false, bS.v, (short)0, accNz[tm][tn], false, false);
            }
        }
        __syncthreads();
    }

    // ---- epilogue: out = mean + bias_mu + eps_out*(noise + bias_sigma) ----
    #pragma unroll
    for (int tn = 0; tn < 4; ++tn) {
        int N = n0 + wn * 64 + tn * 16 + mlo;                 // C/D: N = lane&15
        float bm = bmu[N];
        float bs = bsg[N];
        #pragma unroll
        for (int tm = 0; tm < 2; ++tm) {
            int Mbase = m0 + wm * 32 + tm * 16 + (ksel << 3); // C/D: M = r + 8*(lane>>4)
            #pragma unroll
            for (int r = 0; r < 8; ++r) {
                size_t idx = (size_t)(Mbase + r) * Udim + N;
                float eo = eps_out[idx];
                out[idx] = accMu[tm][tn][r] + bm + eo * (accNz[tm][tn][r] + bs);
            }
        }
    }
}

extern "C" void kernel_launch(void* const* d_in, const int* in_sizes, int n_in,
                              void* d_out, int out_size, void* d_ws, size_t ws_size,
                              hipStream_t stream) {
    const float* x    = (const float*)d_in[0];
    const float* wmu  = (const float*)d_in[1];
    const float* wsg  = (const float*)d_in[2];
    const float* bmu  = (const float*)d_in[3];
    const float* bsg  = (const float*)d_in[4];
    const float* ein  = (const float*)d_in[5];
    const float* eout = (const float*)d_in[6];
    float* out = (float*)d_out;

    const int D = 2048, U = 2048;
    const int B = in_sizes[0] / D;             // 4096

    dim3 grid(U / TILE_N, B / TILE_M);         // (16, 32)
    noisy_dense_wmma<<<grid, 256, 0, stream>>>(x, wmu, wsg, bmu, bsg, ein, eout, out, D, U);
}
